// MultiHeadAttention_52793738002487
// MI455X (gfx1250) — matrix operand order repaired and checked by reading, not compiled, not run on hardware
//
#include <hip/hip_runtime.h>
#include <hip/hip_bf16.h>
#include <math.h>

// ---------------------------------------------------------------------------
// MI455X (gfx1250) fused MHA forward:
//   1) qkv_proj_kernel : x @ {Wq,Wk,Wv}^T + b  -> bf16 head-major [h*B+b,s,64]
//   2) flash_attn_kernel: TDM double-buffered K/V tiles (tensor_load_to_lds,
//      TENSORcnt), streaming QK^T (pre_score f32 via NT stores), mask +
//      online softmax + P@V, all bf16 WMMA / f32 acc, wave32 shuffles.
//   3) out_proj_kernel : heads @ Wo^T + bo -> f32 output
// Store-bound on the mandatory 537MB pre_score stream (~23us @ 23.3TB/s);
// TDM overlaps next-tile K/V fetch with current-tile WMMA/softmax.
// ---------------------------------------------------------------------------

typedef __bf16 bf16;
typedef __attribute__((ext_vector_type(16))) __bf16 v16bf;
typedef __attribute__((ext_vector_type(8)))  float  v8f;
typedef unsigned int u32x4 __attribute__((ext_vector_type(4)));
typedef int          i32x8 __attribute__((ext_vector_type(8)));
typedef int          i32x4 __attribute__((ext_vector_type(4)));

union V16 { v16bf v; unsigned int u[8]; };

__device__ __forceinline__ v8f wmma_bf16(v16bf a, v16bf b, v8f c) {
  // (neg_a, A, neg_b, B, c_mod, C, reuse_a, reuse_b)
  return __builtin_amdgcn_wmma_f32_16x16x32_bf16(false, a, false, b, (short)0, c,
                                                 false, false);
}

// A-operand 16x32 bf16 fragment. lane.M = lane&15, K range split by lane-half:
//   K(r,half,lo) = half*16 + r*2 + lo   (pairs contiguous -> u32)
__device__ __forceinline__ v16bf load_afrag(const bf16* rowp, int half) {
  V16 t;
#pragma unroll
  for (int r = 0; r < 8; ++r) {
    const int k = (half << 4) + (r << 1);
    t.u[r] = *(const unsigned int*)(rowp + k);
  }
  return t.v;
}

// B-operand 32x16 bf16 fragment, tile stored [N][K] (lane's N-row contiguous).
// lane.N = lane&15; K(r,half,lo) = half*16 + r*2 + lo  (same k order as load_afrag)
__device__ __forceinline__ v16bf load_bfrag_rowmajor(const bf16* rowp, int half) {
  V16 t;
#pragma unroll
  for (int r = 0; r < 8; ++r) {
    const int k = (half << 4) + (r << 1);
    t.u[r] = *(const unsigned int*)(rowp + k);
  }
  return t.v;
}

// B-operand 32x16 bf16 fragment, tile stored [K][N]: gather down a column.
__device__ __forceinline__ v16bf load_bfrag_colmajor(const bf16* base, int ldk,
                                                     int half, int ncol) {
  V16 t;
#pragma unroll
  for (int r = 0; r < 8; ++r) {
    const int k = (half << 4) + (r << 1);
    const unsigned short lo =
        __builtin_bit_cast(unsigned short, base[(size_t)k * ldk + ncol]);
    const unsigned short hi =
        __builtin_bit_cast(unsigned short, base[(size_t)(k + 1) * ldk + ncol]);
    t.u[r] = (unsigned int)lo | ((unsigned int)hi << 16);
  }
  return t.v;
}

// ---------------------------------------------------------------------------
// TDM: DMA one 64x64 bf16 tile (row stride 64 elems in a [2048,64] tensor)
// from global into LDS, hardware-padding +16B after each 128B row so the
// LDS image lands in a [64][72] bank-conflict-free layout.  D# fields per
// CDNA5 ISA (group0/group1; groups 2,3 zero for 2-D).
// This toolchain exposes the 6-operand builtin: (g0,g1,g2,g3,g4,cpol).
// ---------------------------------------------------------------------------
__device__ __forceinline__ void tdm_load_tile_64x64(unsigned lds_addr,
                                                    const bf16* gptr) {
  const unsigned long long ga = (unsigned long long)(size_t)gptr;
  u32x4 g0;
  g0[0] = 1u;                                  // count=1 (valid), user mode
  g0[1] = lds_addr;                            // LDS byte address
  g0[2] = (unsigned)ga;                        // global_addr[31:0]
  g0[3] = (unsigned)(ga >> 32) | (2u << 30);   // global_addr[56:32] | type=2
  i32x8 g1;
  g1[0] = (1 << 16)        // data_size = 2 bytes
        | (1 << 20)        // pad_enable (LDS dest padding)
        | (4 << 22)        // pad_interval: 32 DWORDs (=128B, one tile row)
        | (3 << 25);       // pad_amount : 4 DWORDs  (=16B -> 144B row stride)
  g1[1] = (64 << 16);      // tensor_dim0 = 64 elems (bits 79:48, low half)
  g1[2] = (2048 << 16);    // tensor_dim1 = 2048     (bits 111:80, low half)
  g1[3] = (64 << 16);      // tile_dim0 = 64         (bits 127:112)
  g1[4] = 64;              // tile_dim1 = 64, tile_dim2 = 0
  g1[5] = 64;              // tensor_dim0_stride = 64 elems (bits 207:160)
  g1[6] = 0;               // tensor_dim1_stride = 0 (unused for 2-D tile)
  g1[7] = 0;
  const i32x4 gz4 = {0, 0, 0, 0};              // groups 2/3: NULL (<=2-D)
  const i32x8 gz8 = {0, 0, 0, 0, 0, 0, 0, 0};  // unused trailing group
  __builtin_amdgcn_tensor_load_to_lds(g0, g1, gz4, gz4, gz8, 0);
}

// ---------------------------------------------------------------------------
// Kernel 1: QKV projection.  [8192,512] x [512,512]^T, bf16 WMMA, f32 acc.
// Block tile 128x128, 8 waves (4 M x 2 N), each wave 32x64 (2x4 fragments).
// grid.z selects Q / K / V.  Output scattered to head-major bf16.
// ---------------------------------------------------------------------------
__global__ __launch_bounds__(256) void qkv_proj_kernel(
    const float* __restrict__ x,
    const float* __restrict__ Wq, const float* __restrict__ bq,
    const float* __restrict__ Wk, const float* __restrict__ bk,
    const float* __restrict__ Wv, const float* __restrict__ bv,
    bf16* __restrict__ Qb, bf16* __restrict__ Kb, bf16* __restrict__ Vb) {
  const float* W; const float* bias; bf16* out;
  if (blockIdx.z == 0)      { W = Wq; bias = bq; out = Qb; }
  else if (blockIdx.z == 1) { W = Wk; bias = bk; out = Kb; }
  else                      { W = Wv; bias = bv; out = Vb; }

  __shared__ bf16 As[128][40];   // 40-elem stride: conflict-free frag reads
  __shared__ bf16 Bs[128][40];

  const int tid  = threadIdx.x;
  const int wave = tid >> 5, lane = tid & 31;
  const int half = lane >> 4, nlo = lane & 15;
  const int wm = wave & 3, wn = wave >> 2;
  const int m0 = blockIdx.x * 128;
  const int n0 = blockIdx.y * 128;

  v8f acc[2][4];
  const v8f zero8 = {0.f, 0.f, 0.f, 0.f, 0.f, 0.f, 0.f, 0.f};
#pragma unroll
  for (int mi = 0; mi < 2; ++mi)
#pragma unroll
    for (int f = 0; f < 4; ++f) acc[mi][f] = zero8;

  const int li = tid >> 1, lc = (tid & 1) * 16;
  for (int k0 = 0; k0 < 512; k0 += 32) {
    __syncthreads();
    {
      const float* src = x + (size_t)(m0 + li) * 512 + k0 + lc;
#pragma unroll
      for (int t = 0; t < 16; ++t) As[li][lc + t] = (bf16)src[t];
      const float* wsrc = W + (size_t)(n0 + li) * 512 + k0 + lc;
#pragma unroll
      for (int t = 0; t < 16; ++t) Bs[li][lc + t] = (bf16)wsrc[t];
      if (k0 + 32 < 512) {                     // -> global_prefetch_b8
        __builtin_prefetch(src + 32, 0, 1);
        __builtin_prefetch(wsrc + 32, 0, 1);
      }
    }
    __syncthreads();
    v16bf a2[2], b4[4];
#pragma unroll
    for (int mi = 0; mi < 2; ++mi)
      a2[mi] = load_afrag(&As[wm * 32 + mi * 16 + nlo][0], half);
#pragma unroll
    for (int f = 0; f < 4; ++f)
      b4[f] = load_bfrag_rowmajor(&Bs[wn * 64 + f * 16 + nlo][0], half);
#pragma unroll
    for (int mi = 0; mi < 2; ++mi)
#pragma unroll
      for (int f = 0; f < 4; ++f)
        acc[mi][f] = wmma_bf16(a2[mi], b4[f], acc[mi][f]);
  }

  // bias + bf16 convert + scatter to head-major [h*B+b, s, dh]
#pragma unroll
  for (int mi = 0; mi < 2; ++mi)
#pragma unroll
    for (int f = 0; f < 4; ++f) {
      const int c = n0 + wn * 64 + f * 16 + nlo;
      const float bc = bias[c];
      const int h = c >> 6, d = c & 63;
#pragma unroll
      for (int r = 0; r < 8; ++r) {
        const int m = m0 + wm * 32 + mi * 16 + half * 8 + r;
        const int b = m >> 11, s = m & 2047;
        out[(((size_t)h * 4 + b) * 2048 + s) * 64 + d] = (bf16)(acc[mi][f][r] + bc);
      }
    }
}

// ---------------------------------------------------------------------------
// Kernel 2: flash attention + full pre_score emission.
// Block = 128 query rows of one n = h*B+b.  Each of 8 waves owns 16 rows.
// K/V 64-key tiles are TDM-DMA'd into double-buffered LDS (wave 0 issues,
// TENSORcnt drains the oldest pair); next-tile DMA overlaps current compute.
// Row stats reduce across the 16 lanes of a lane-half via shfl_xor {1,2,4,8}.
// ---------------------------------------------------------------------------
__global__ __launch_bounds__(256) void flash_attn_kernel(
    const bf16* __restrict__ Qb, const bf16* __restrict__ Kb,
    const bf16* __restrict__ Vb, const int* __restrict__ pad_mask,
    bf16* __restrict__ Ob, float* __restrict__ pre_score) {
  __shared__ bf16 Kt[2][64][72];     // [buf][t][dh], 72-stride (TDM pad)
  __shared__ bf16 Vt[2][64][72];
  __shared__ bf16 Ps[8][16][72];     // wave-private P staging (C->A relayout)

  const int tid  = threadIdx.x;
  const int wave = tid >> 5, lane = tid & 31;
  const int half = lane >> 4, nlo = lane & 15;
  const int n     = blockIdx.y;      // head-major: n = h*B + b
  const int pad_b = n >> 3;          // reference quirk: padding uses n / H
  const int row0  = blockIdx.x * 128 + wave * 16;

  const bf16* Qn = Qb + (size_t)n * 2048 * 64;
  const bf16* Kn = Kb + (size_t)n * 2048 * 64;
  const bf16* Vn = Vb + (size_t)n * 2048 * 64;
  float* psn = pre_score + (size_t)n * 2048 * 2048;

  v16bf aQ[2];
#pragma unroll
  for (int ks = 0; ks < 2; ++ks)
    aQ[ks] = load_afrag(Qn + (size_t)(row0 + nlo) * 64 + ks * 32, half);

  const v8f zero8 = {0.f, 0.f, 0.f, 0.f, 0.f, 0.f, 0.f, 0.f};
  v8f Oacc[4];
#pragma unroll
  for (int f = 0; f < 4; ++f) Oacc[f] = zero8;
  float m[8], ssum[8];
#pragma unroll
  for (int r = 0; r < 8; ++r) { m[r] = -3.4e38f; ssum[r] = 0.f; }

  // Prologue: wave 0 DMAs tile 0 into buffer 0 (TENSORcnt += 2).
  if (wave == 0) {
    tdm_load_tile_64x64((unsigned)(size_t)&Kt[0][0][0], Kn);
    tdm_load_tile_64x64((unsigned)(size_t)&Vt[0][0][0], Vn);
  }

  for (int j0 = 0; j0 < 2048; j0 += 64) {
    const int buf = (j0 >> 6) & 1;
    __syncthreads();   // all waves done reading buf^1 (previous tile)
    if (wave == 0) {
      if (j0 + 64 < 2048) {  // DMA next tile into the other buffer
        tdm_load_tile_64x64((unsigned)(size_t)&Kt[buf ^ 1][0][0],
                            Kn + (size_t)(j0 + 64) * 64);
        tdm_load_tile_64x64((unsigned)(size_t)&Vt[buf ^ 1][0][0],
                            Vn + (size_t)(j0 + 64) * 64);
        __builtin_amdgcn_s_wait_tensorcnt(2);  // drain current tile's pair
      } else {
        __builtin_amdgcn_s_wait_tensorcnt(0);
      }
    }
    __syncthreads();   // tile j0 visible to all waves

    // S = Q K^T  (16 rows x 64 cols per wave; dh contraction = 2 WMMA steps)
    v8f sc[4];
#pragma unroll
    for (int f = 0; f < 4; ++f) {
      sc[f] = zero8;
#pragma unroll
      for (int ks = 0; ks < 2; ++ks) {
        const v16bf bK =
            load_bfrag_rowmajor(&Kt[buf][f * 16 + nlo][ks * 32], half);
        sc[f] = wmma_bf16(aQ[ks], bK, sc[f]);
      }
    }

    // scale, stream unmasked pre_score (non-temporal), then mask in-register
#pragma unroll
    for (int f = 0; f < 4; ++f) {
      const int col = j0 + f * 16 + nlo;
      const int pmv = pad_mask[pad_b * 2048 + col];
#pragma unroll
      for (int r = 0; r < 8; ++r) {
        const int row = row0 + half * 8 + r;
        float v = sc[f][r] * 0.125f;                       // 1/sqrt(64)
        __builtin_nontemporal_store(v, &psn[(size_t)row * 2048 + col]);
        if (col > row) v = -4294967295.0f;                 // causal
        if (pmv == 0)  v = -4294967295.0f;                 // padding quirk
        sc[f][r] = v;
      }
    }

    // online softmax
    float mnew[8], alpha[8];
#pragma unroll
    for (int r = 0; r < 8; ++r) {
      float lm = fmaxf(fmaxf(sc[0][r], sc[1][r]), fmaxf(sc[2][r], sc[3][r]));
      lm = fmaxf(lm, __shfl_xor(lm, 1, 32));
      lm = fmaxf(lm, __shfl_xor(lm, 2, 32));
      lm = fmaxf(lm, __shfl_xor(lm, 4, 32));
      lm = fmaxf(lm, __shfl_xor(lm, 8, 32));
      mnew[r]  = fmaxf(m[r], lm);
      alpha[r] = __expf(m[r] - mnew[r]);
      m[r] = mnew[r];
    }
#pragma unroll
    for (int f = 0; f < 4; ++f)
#pragma unroll
      for (int r = 0; r < 8; ++r) sc[f][r] = __expf(sc[f][r] - mnew[r]);
#pragma unroll
    for (int r = 0; r < 8; ++r) {
      float s = (sc[0][r] + sc[1][r]) + (sc[2][r] + sc[3][r]);
      s += __shfl_xor(s, 1, 32);
      s += __shfl_xor(s, 2, 32);
      s += __shfl_xor(s, 4, 32);
      s += __shfl_xor(s, 8, 32);
      ssum[r] = ssum[r] * alpha[r] + s;
    }
#pragma unroll
    for (int f = 0; f < 4; ++f)
#pragma unroll
      for (int r = 0; r < 8; ++r) Oacc[f][r] *= alpha[r];

    // P: C-layout regs -> wave-private LDS -> A-layout fragments
#pragma unroll
    for (int f = 0; f < 4; ++f)
#pragma unroll
      for (int r = 0; r < 8; ++r)
        Ps[wave][half * 8 + r][f * 16 + nlo] = (bf16)sc[f][r];
    asm volatile("s_wait_dscnt 0" ::: "memory");
    v16bf aP[2];
#pragma unroll
    for (int ks = 0; ks < 2; ++ks)
      aP[ks] = load_afrag(&Ps[wave][nlo][ks * 32], half);

    // O += P V
#pragma unroll
    for (int f = 0; f < 4; ++f)
#pragma unroll
      for (int ks = 0; ks < 2; ++ks) {
        const v16bf bV =
            load_bfrag_colmajor(&Vt[buf][ks * 32][0], 72, half, f * 16 + nlo);
        Oacc[f] = wmma_bf16(aP[ks], bV, Oacc[f]);
      }
  }

  bf16* On = Ob + (size_t)n * 2048 * 64;
#pragma unroll
  for (int f = 0; f < 4; ++f)
#pragma unroll
    for (int r = 0; r < 8; ++r) {
      const int row = row0 + half * 8 + r;
      On[(size_t)row * 64 + f * 16 + nlo] = (bf16)(Oacc[f][r] / ssum[r]);
    }
}

// ---------------------------------------------------------------------------
// Kernel 3: output projection.  heads(bf16, gathered head-major) @ Wo^T + bo.
// Same 128x128 tiling as kernel 1; f32 output to d_out.
// ---------------------------------------------------------------------------
__global__ __launch_bounds__(256) void out_proj_kernel(
    const bf16* __restrict__ Ob, const float* __restrict__ Wo,
    const float* __restrict__ bo, float* __restrict__ outp) {
  __shared__ bf16 As[128][40];
  __shared__ bf16 Bs[128][40];

  const int tid  = threadIdx.x;
  const int wave = tid >> 5, lane = tid & 31;
  const int half = lane >> 4, nlo = lane & 15;
  const int wm = wave & 3, wn = wave >> 2;
  const int m0 = blockIdx.x * 128;
  const int n0 = blockIdx.y * 128;

  v8f acc[2][4];
  const v8f zero8 = {0.f, 0.f, 0.f, 0.f, 0.f, 0.f, 0.f, 0.f};
#pragma unroll
  for (int mi = 0; mi < 2; ++mi)
#pragma unroll
    for (int f = 0; f < 4; ++f) acc[mi][f] = zero8;

  const int li = tid >> 1, lc = (tid & 1) * 16;
  const int mrow_l = m0 + li;
  const int bb = mrow_l >> 11, ss = mrow_l & 2047;

  for (int k0 = 0; k0 < 512; k0 += 32) {
    __syncthreads();
    {
      const int h = k0 >> 6, dbase = k0 & 63;   // heads[.,h*64+d] = Ob[h*B+b][s][d]
      const bf16* src = Ob + (((size_t)h * 4 + bb) * 2048 + ss) * 64 + dbase + lc;
      const uint4 p0 = *(const uint4*)(src);
      const uint4 p1 = *(const uint4*)(src + 8);
      uint2* ad = (uint2*)&As[li][lc];
      ad[0] = make_uint2(p0.x, p0.y); ad[1] = make_uint2(p0.z, p0.w);
      ad[2] = make_uint2(p1.x, p1.y); ad[3] = make_uint2(p1.z, p1.w);
      const float* wsrc = Wo + (size_t)(n0 + li) * 512 + k0 + lc;
#pragma unroll
      for (int t = 0; t < 16; ++t) Bs[li][lc + t] = (bf16)wsrc[t];
      if (k0 + 32 < 512) __builtin_prefetch(wsrc + 32, 0, 1);
    }
    __syncthreads();
    v16bf a2[2], b4[4];
#pragma unroll
    for (int mi = 0; mi < 2; ++mi)
      a2[mi] = load_afrag(&As[wm * 32 + mi * 16 + nlo][0], half);
#pragma unroll
    for (int f = 0; f < 4; ++f)
      b4[f] = load_bfrag_rowmajor(&Bs[wn * 64 + f * 16 + nlo][0], half);
#pragma unroll
    for (int mi = 0; mi < 2; ++mi)
#pragma unroll
      for (int f = 0; f < 4; ++f)
        acc[mi][f] = wmma_bf16(a2[mi], b4[f], acc[mi][f]);
  }

#pragma unroll
  for (int mi = 0; mi < 2; ++mi)
#pragma unroll
    for (int f = 0; f < 4; ++f) {
      const int c = n0 + wn * 64 + f * 16 + nlo;
      const float bc = bo[c];
#pragma unroll
      for (int r = 0; r < 8; ++r) {
        const int mr = m0 + wm * 32 + mi * 16 + half * 8 + r;
        outp[(size_t)mr * 512 + c] = acc[mi][f][r] + bc;
      }
    }
}

// ---------------------------------------------------------------------------
extern "C" void kernel_launch(void* const* d_in, const int* in_sizes, int n_in,
                              void* d_out, int out_size, void* d_ws, size_t ws_size,
                              hipStream_t stream) {
  (void)in_sizes; (void)n_in; (void)out_size; (void)ws_size;
  const float* x  = (const float*)d_in[0];
  const int*   pm = (const int*)d_in[1];
  const float* Wq = (const float*)d_in[2];
  const float* bq = (const float*)d_in[3];
  const float* Wk = (const float*)d_in[4];
  const float* bk = (const float*)d_in[5];
  const float* Wv = (const float*)d_in[6];
  const float* bv = (const float*)d_in[7];
  const float* Wo = (const float*)d_in[8];
  const float* bo = (const float*)d_in[9];

  float* attn_out  = (float*)d_out;                         // [4,2048,512]
  float* pre_score = attn_out + (size_t)4 * 2048 * 512;     // [32,2048,2048]

  const size_t headElems = (size_t)32 * 2048 * 64;          // per Q/K/V/O buffer
  bf16* Qb = (bf16*)d_ws;                                   // ws: 4 x 8MB = 32MB
  bf16* Kb = Qb + headElems;
  bf16* Vb = Kb + headElems;
  bf16* Ob = Vb + headElems;

  qkv_proj_kernel<<<dim3(64, 4, 3), 256, 0, stream>>>(x, Wq, bq, Wk, bk, Wv, bv,
                                                      Qb, Kb, Vb);
  flash_attn_kernel<<<dim3(16, 32), 256, 0, stream>>>(Qb, Kb, Vb, pm, Ob, pre_score);
  out_proj_kernel<<<dim3(64, 4), 256, 0, stream>>>(Ob, Wo, bo, attn_out);
}
